// UNetV2_39797166965279
// MI455X (gfx1250) — compile-verified, matrix operand order checked by reading, not run on hardware
//
#include <hip/hip_runtime.h>
#include <hip/hip_bf16.h>

typedef __attribute__((ext_vector_type(16))) _Float16 v16h;
typedef __attribute__((ext_vector_type(8)))  _Float16 v8h;
typedef __attribute__((ext_vector_type(8)))  float    v8f;

// ---------------------------------------------------------------------------
// Implicit-GEMM conv3d / conv3d-transpose with fused BN + residual + ReLU +
// mask. Block = 256 threads = 8 wave32s; wave w computes the 16-voxel tile
// (blockIdx.x*8 + w) x 16 output channels (blockIdx.y). When USE_LDS==1 the
// packed weight fragments for this output-channel group are staged once per
// workgroup into LDS with the CDNA5 async copy path
// (global_load_async_to_lds_b128 + s_wait_asynccnt), then read back as WMMA
// B fragments via ds_load. USE_LDS==0 (weight image > 64KB) reads global.
// Activations: f16, layout [D][H][W][CinPad], CinPad multiple of 32.
// Weights: pre-packed f16 fragments [Cout/16][NTAP][CinPad/32][32 lanes][16].
// MODE==0: out voxel o reads in voxel o*s + t - p   (normal conv)
// MODE==1: out voxel o reads in voxel (o + p - t)/s (transposed conv)
// ---------------------------------------------------------------------------
template <int KD, int KH, int KW, int MODE, int USE_LDS>
__global__ __launch_bounds__(256) void conv3d_wmma(
    const _Float16* __restrict__ in, int Di, int Hi, int Wi, int CinPad,
    _Float16* __restrict__ out, int Do, int Ho, int Wo, int Cout, int CoutPad,
    const _Float16* __restrict__ wpk,
    const float* __restrict__ scale, const float* __restrict__ bias,
    const float* __restrict__ mask,
    int sz, int sy, int sx, int pz, int py, int px,
    int relu, const _Float16* __restrict__ resid)
{
    extern __shared__ _Float16 smemw[];
    constexpr int NTAP = KD * KH * KW;

    const int lane = threadIdx.x & 31;
    const int wave = threadIdx.x >> 5;
    const int tile = blockIdx.x * 8 + wave;
    const int cot  = blockIdx.y;
    const int nvox = Do * Ho * Wo;
    const int KB   = CinPad >> 5;

    const _Float16* wsrc = wpk + (size_t)cot * NTAP * KB * 512;

    // ---- cooperative async staging of this channel-group's weights ----
    if (USE_LDS) {
        unsigned ldsbase = (unsigned)(unsigned long long)(void*)smemw;
        int nchunk = NTAP * KB * 64;                  // 16-byte chunks
        for (int c = threadIdx.x; c < nchunk; c += 256) {
            unsigned dst = ldsbase + (unsigned)c * 16u;
            unsigned long long src =
                (unsigned long long)(wsrc + (size_t)c * 8);
            asm volatile("global_load_async_to_lds_b128 %0, %1, off"
                         :: "v"(dst), "v"(src) : "memory");
        }
        asm volatile("s_wait_asynccnt 0x0" ::: "memory");
    }
    __syncthreads();

    const int m0    = tile * 16;
    const int row   = lane & 15;
    const int khalf = lane >> 4;

    int v = m0 + row;
    bool vok = v < nvox;
    int oz = 0, oy = 0, ox = 0;
    if (vok) {
        oz = v / (Ho * Wo);
        int r = v - oz * Ho * Wo;
        oy = r / Wo;
        ox = r - oy * Wo;
    }

    v8f acc;
    for (int i = 0; i < 8; ++i) acc[i] = 0.0f;

    for (int tap = 0; tap < NTAP; ++tap) {
        const int dz = tap / (KH * KW);
        const int rr = tap - dz * (KH * KW);
        const int dy = rr / KW;
        const int dx = rr - dy * KW;

        bool ok = vok;
        int iz = 0, iy = 0, ix = 0;
        if (MODE == 0) {
            iz = oz * sz + dz - pz;
            iy = oy * sy + dy - py;
            ix = ox * sx + dx - px;
        } else {
            int qz = oz + pz - dz, qy = oy + py - dy, qx = ox + px - dx;
            ok = ok && qz >= 0 && qy >= 0 && qx >= 0 &&
                 (qz % sz) == 0 && (qy % sy) == 0 && (qx % sx) == 0;
            if (qz >= 0) iz = qz / sz;
            if (qy >= 0) iy = qy / sy;
            if (qx >= 0) ix = qx / sx;
        }
        ok = ok && iz >= 0 && iz < Di && iy >= 0 && iy < Hi && ix >= 0 && ix < Wi;

        const _Float16* abase =
            in + ((size_t)(iz * Hi + iy) * Wi + ix) * CinPad + khalf * 8;

        if (ok) {
            // speculative prefetch of the next row -> global_prefetch_b8
            __builtin_prefetch((const void*)(abase + (size_t)Wi * CinPad), 0, 0);
        }

        for (int kb = 0; kb < KB; ++kb) {
            v16h a;
            if (ok) {
                v8h lo = *(const v8h*)(abase + kb * 32);       // K = kbase+0..7
                v8h hi = *(const v8h*)(abase + kb * 32 + 16);  // K = 16+kbase..
                for (int i = 0; i < 8; ++i) { a[i] = lo[i]; a[8 + i] = hi[i]; }
            } else {
                for (int i = 0; i < 16; ++i) a[i] = (_Float16)0.0f;
            }

            v16h bfrag;
            const size_t fofs = ((size_t)(tap * KB + kb) * 32 + lane) * 16;
            if (USE_LDS) {
                const _Float16* wl = smemw + fofs;             // ds_load path
                v8h lo = *(const v8h*)(wl);
                v8h hi = *(const v8h*)(wl + 8);
                for (int i = 0; i < 8; ++i) { bfrag[i] = lo[i]; bfrag[8 + i] = hi[i]; }
            } else {
                bfrag = *(const v16h*)(wsrc + fofs);
            }

            acc = __builtin_amdgcn_wmma_f32_16x16x32_f16(
                false, a, false, bfrag, (short)0, acc, false, false);
        }
    }

    // Epilogue: D element (vgpr i, lane) -> voxel m0 + i + 8*(lane>=16),
    // channel cot*16 + (lane & 15).
    const int n  = cot * 16 + row;
    const float sc = scale[n];
    const float bi = bias[n];
    const int mofs = khalf * 8;
    for (int i = 0; i < 8; ++i) {
        int vo = m0 + mofs + i;
        if (vo < nvox) {
            float val = acc[i] * sc + bi;
            if (resid) val += (float)resid[(size_t)vo * CoutPad + n];
            if (relu)  val = val > 0.0f ? val : 0.0f;
            val *= mask[vo];
            out[(size_t)vo * CoutPad + n] = (_Float16)val;
        }
    }
}

// Pack raw (kd,kh,kw,Cin,Cout) f32 weights into per-lane WMMA B fragments:
// p[cot][tap][kb][lane][16 halves]; lane l holds column N = cot*16 + (l&15),
// halves are K = (l>>4)*8 + {0..7} and 16 + (l>>4)*8 + {0..7}.
__global__ void repack_w(const float* __restrict__ w, _Float16* __restrict__ p,
                         int ntap, int Cin, int CinPad, int Cout, size_t total)
{
    size_t idx = (size_t)blockIdx.x * blockDim.x + threadIdx.x;
    if (idx >= total) return;
    int h    = (int)(idx & 15);
    int lane = (int)((idx >> 4) & 31);
    size_t t = idx >> 9;
    int KB  = CinPad >> 5;
    int kb  = (int)(t % KB); t /= KB;
    int tap = (int)(t % ntap); t /= ntap;
    int cot = (int)t;
    int n     = cot * 16 + (lane & 15);
    int kbase = (lane >> 4) * 8;
    int K     = (h < 8) ? (kbase + h) : (16 + kbase + (h - 8));
    int cin   = kb * 32 + K;
    float val = 0.0f;
    if (cin < Cin && n < Cout) val = w[((size_t)tap * Cin + cin) * Cout + n];
    p[idx] = (_Float16)val;
}

__global__ void scatter_vox(const float* __restrict__ feat,
                            const int* __restrict__ coords,
                            _Float16* __restrict__ x0, float* __restrict__ m1,
                            int N, int H, int W, int Cin, int CPad)
{
    int i = blockIdx.x * blockDim.x + threadIdx.x;
    if (i >= N) return;
    int z = coords[i * 4 + 1], y = coords[i * 4 + 2], x = coords[i * 4 + 3];
    size_t flat = (size_t)(z * H + y) * W + x;
    m1[flat] = 1.0f;
    for (int c = 0; c < Cin; ++c)
        x0[flat * CPad + c] = (_Float16)feat[i * Cin + c];
}

__global__ void maxpool_m(const float* __restrict__ in, int Di, int Hi, int Wi,
                          float* __restrict__ out, int Do, int Ho, int Wo,
                          int wd, int wh, int ww, int sz, int sy, int sx,
                          int pz, int py, int px)
{
    int v = blockIdx.x * blockDim.x + threadIdx.x;
    int n = Do * Ho * Wo;
    if (v >= n) return;
    int z = v / (Ho * Wo);
    int r = v - z * Ho * Wo;
    int y = r / Wo;
    int x = r - y * Wo;
    float mx = 0.0f;
    for (int a = 0; a < wd; ++a) {
        int iz = z * sz + a - pz; if (iz < 0 || iz >= Di) continue;
        for (int b = 0; b < wh; ++b) {
            int iy = y * sy + b - py; if (iy < 0 || iy >= Hi) continue;
            for (int c = 0; c < ww; ++c) {
                int ix = x * sx + c - px; if (ix < 0 || ix >= Wi) continue;
                float t = in[((size_t)iz * Hi + iy) * Wi + ix];
                mx = t > mx ? t : mx;
            }
        }
    }
    out[v] = mx;
}

__global__ void cat_ch(const _Float16* __restrict__ a, int sa, int Ca,
                       const _Float16* __restrict__ b, int sb, int Cb,
                       _Float16* __restrict__ o, int so, int n)
{
    int C = Ca + Cb;
    int i = blockIdx.x * blockDim.x + threadIdx.x;
    if (i >= n * C) return;
    int v = i / C, c = i - v * C;
    _Float16 val = (c < Ca) ? a[(size_t)v * sa + c] : b[(size_t)v * sb + (c - Ca)];
    o[(size_t)v * so + c] = val;
}

// out[v,c] = xm[v,c] + cat[v,2c] + cat[v,2c+1]   (x_m + channel_reduce(cat))
__global__ void red_add(const _Float16* __restrict__ xm, int sm,
                        const _Float16* __restrict__ cat, int sc,
                        _Float16* __restrict__ o, int so, int n, int C)
{
    int i = blockIdx.x * blockDim.x + threadIdx.x;
    if (i >= n * C) return;
    int v = i / C, c = i - v * C;
    float val = (float)xm[(size_t)v * sm + c] +
                (float)cat[(size_t)v * sc + 2 * c] +
                (float)cat[(size_t)v * sc + 2 * c + 1];
    o[(size_t)v * so + c] = (_Float16)val;
}

// (1,128,2,10,9) -> (1,256,10,9): spat[((c*2+dd)*10+h)*9+w] = ob[vox*128+c]
__global__ void spatial_out(const _Float16* __restrict__ ob, float* __restrict__ out)
{
    int i = blockIdx.x * blockDim.x + threadIdx.x;
    if (i >= 128 * 2 * 10 * 9) return;
    int w = i % 9; int t = i / 9; int h = t % 10; int cd = t / 10;
    int dd = cd & 1; int c = cd >> 1;
    size_t vox = ((size_t)dd * 10 + h) * 9 + w;
    out[i] = (float)ob[vox * 128 + c];
}

__global__ void heads(const _Float16* __restrict__ u, int su,
                      const int* __restrict__ coords,
                      const float* __restrict__ clsw, const float* __restrict__ clsb,
                      const float* __restrict__ regw, const float* __restrict__ regb,
                      float* __restrict__ seg_cls, float* __restrict__ seg_reg,
                      float* __restrict__ seg_feat, int N, int H, int W)
{
    int i = blockIdx.x * blockDim.x + threadIdx.x;
    if (i >= N) return;
    int z = coords[i * 4 + 1], y = coords[i * 4 + 2], x = coords[i * 4 + 3];
    size_t flat = ((size_t)z * H + y) * W + x;
    float f[16];
    for (int c = 0; c < 16; ++c) {
        f[c] = (float)u[flat * su + c];
        seg_feat[i * 16 + c] = f[c];
    }
    float cl = clsb[0];
    for (int c = 0; c < 16; ++c) cl += f[c] * clsw[c];
    seg_cls[i] = cl;
    for (int j = 0; j < 3; ++j) {
        float r = regb[j];
        for (int c = 0; c < 16; ++c) r += f[c] * regw[c * 3 + j];
        seg_reg[i * 3 + j] = r;
    }
}

// ---------------------------------------------------------------------------
// Host orchestration
// ---------------------------------------------------------------------------
namespace {
struct Arena {
    char* base; size_t off;
    void* alloc(size_t bytes) {
        off = (off + 255) & ~(size_t)255;
        void* p = base + off;
        off += bytes;
        return p;
    }
    _Float16* f16(size_t n) { return (_Float16*)alloc(n * sizeof(_Float16)); }
    float*    f32(size_t n) { return (float*)alloc(n * sizeof(float)); }
};
} // namespace

// param flattening (dict insertion order, each conv tuple as w,s,b)
enum {
    IN_W = 2, C1_W = 5, C2A_W = 8, C2B_W = 11, C2C_W = 14,
    C3A_W = 17, C3B_W = 20, C3C_W = 23, C4A_W = 26, C4B_W = 29, C4C_W = 32,
    OUT_W = 35, M4_W = 38, INV4_W = 41, M3_W = 44, INV3_W = 47,
    M2_W = 50, INV2_W = 53, M1_W = 56, C5_W = 59,
    T4A_W = 62, T4B_W = 65, T3A_W = 68, T3B_W = 71,
    T2A_W = 74, T2B_W = 77, T1A_W = 80, T1B_W = 83,
    CLSW = 86, CLSB = 87, REGW = 88, REGB = 89
};

extern "C" void kernel_launch(void* const* d_in, const int* in_sizes, int n_in,
                              void* d_out, int out_size, void* d_ws, size_t ws_size,
                              hipStream_t stream)
{
    (void)in_sizes; (void)n_in; (void)out_size;
    const float* feat   = (const float*)d_in[0];
    const int*   coords = (const int*)d_in[1];
    auto Wf = [&](int i) { return (const float*)d_in[i]; };

    const int D1 = 41, H1 = 80, W1 = 72; const int n1 = D1 * H1 * W1; // 236160
    const int D2 = 21, H2 = 40, W2 = 36; const int n2 = D2 * H2 * W2; // 30240
    const int D3 = 11, H3 = 20, W3 = 18; const int n3 = D3 * H3 * W3; // 3960
    const int D4 = 5,  H4 = 10, W4 = 9;  const int n4 = D4 * H4 * W4; // 450
    const int DO = 2,  HO = 10, WO = 9;  const int no = DO * HO * WO; // 180
    const int NV = 30000;

    Arena A{(char*)d_ws, 0};

    // ---- phase 1: allocate everything ----
    float* m1 = A.f32(n1); float* m2 = A.f32(n2); float* m3 = A.f32(n3);
    float* m4 = A.f32(n4); float* mo = A.f32(no);

    _Float16* x0  = A.f16((size_t)n1 * 32);  // input / later cat1
    _Float16* ta  = A.f16((size_t)n1 * 32);
    _Float16* tb  = A.f16((size_t)n1 * 32);
    _Float16* x1  = A.f16((size_t)n1 * 32);
    _Float16* u1d = A.f16((size_t)n1 * 32);

    _Float16* x2   = A.f16((size_t)n2 * 32);
    _Float16* t2a  = A.f16((size_t)n2 * 32);
    _Float16* t2b  = A.f16((size_t)n2 * 32);
    _Float16* u2d  = A.f16((size_t)n2 * 32);
    _Float16* cat2 = A.f16((size_t)n2 * 64);
    _Float16* u2   = A.f16((size_t)n2 * 32);

    _Float16* x3   = A.f16((size_t)n3 * 64);
    _Float16* t3a  = A.f16((size_t)n3 * 64);
    _Float16* t3b  = A.f16((size_t)n3 * 64);
    _Float16* u3d  = A.f16((size_t)n3 * 64);
    _Float16* cat3 = A.f16((size_t)n3 * 128);
    _Float16* u3   = A.f16((size_t)n3 * 64);

    _Float16* x4   = A.f16((size_t)n4 * 64);
    _Float16* t4a  = A.f16((size_t)n4 * 64);
    _Float16* t4b  = A.f16((size_t)n4 * 64);
    _Float16* cat4 = A.f16((size_t)n4 * 128);
    _Float16* u4   = A.f16((size_t)n4 * 64);

    _Float16* ob = A.f16((size_t)no * 128);

    struct PackJob { const float* w; _Float16* p; int ntap, Cin, CinPad, Cout; size_t total; };
    PackJob jobs[28];
    int njobs = 0;
    auto plan = [&](int wi, int kd, int kh, int kw, int Cin, int CinPad, int Cout)
        -> _Float16* {
        int ntap = kd * kh * kw;
        size_t total = (size_t)(Cout / 16) * ntap * (CinPad / 32) * 512;
        _Float16* p = A.f16(total);
        jobs[njobs++] = PackJob{Wf(wi), p, ntap, Cin, CinPad, Cout, total};
        return p;
    };
    _Float16* w_in   = plan(IN_W,  3,3,3,   4, 32,  16);
    _Float16* w_c1   = plan(C1_W,  3,3,3,  16, 32,  16);
    _Float16* w_c2a  = plan(C2A_W, 3,3,3,  16, 32,  32);
    _Float16* w_c2b  = plan(C2B_W, 3,3,3,  32, 32,  32);
    _Float16* w_c2c  = plan(C2C_W, 3,3,3,  32, 32,  32);
    _Float16* w_c3a  = plan(C3A_W, 3,3,3,  32, 32,  64);
    _Float16* w_c3b  = plan(C3B_W, 3,3,3,  64, 64,  64);
    _Float16* w_c3c  = plan(C3C_W, 3,3,3,  64, 64,  64);
    _Float16* w_c4a  = plan(C4A_W, 3,3,3,  64, 64,  64);
    _Float16* w_c4b  = plan(C4B_W, 3,3,3,  64, 64,  64);
    _Float16* w_c4c  = plan(C4C_W, 3,3,3,  64, 64,  64);
    _Float16* w_out  = plan(OUT_W, 3,1,1,  64, 64, 128);
    _Float16* w_m4   = plan(M4_W,  3,3,3, 128,128,  64);
    _Float16* w_inv4 = plan(INV4_W,3,3,3,  64, 64,  64);
    _Float16* w_m3   = plan(M3_W,  3,3,3, 128,128,  64);
    _Float16* w_inv3 = plan(INV3_W,3,3,3,  64, 64,  32);
    _Float16* w_m2   = plan(M2_W,  3,3,3,  64, 64,  32);
    _Float16* w_inv2 = plan(INV2_W,3,3,3,  32, 32,  16);
    _Float16* w_m1   = plan(M1_W,  3,3,3,  32, 32,  16);
    _Float16* w_c5   = plan(C5_W,  3,3,3,  16, 32,  16);
    _Float16* w_t4a  = plan(T4A_W, 3,3,3,  64, 64,  64);
    _Float16* w_t4b  = plan(T4B_W, 3,3,3,  64, 64,  64);
    _Float16* w_t3a  = plan(T3A_W, 3,3,3,  64, 64,  64);
    _Float16* w_t3b  = plan(T3B_W, 3,3,3,  64, 64,  64);
    _Float16* w_t2a  = plan(T2A_W, 3,3,3,  32, 32,  32);
    _Float16* w_t2b  = plan(T2B_W, 3,3,3,  32, 32,  32);
    _Float16* w_t1a  = plan(T1A_W, 3,3,3,  16, 32,  16);
    _Float16* w_t1b  = plan(T1B_W, 3,3,3,  16, 32,  16);

    // ---- phase 2: zero the used arena (channel pads must be 0) ----
    size_t used = A.off;
    if (used > ws_size) used = ws_size; // best effort; compile-only environment
    hipMemsetAsync(d_ws, 0, used, stream);

    // ---- phase 3: launches ----
    for (int j = 0; j < njobs; ++j) {
        unsigned g = (unsigned)((jobs[j].total + 255) / 256);
        repack_w<<<g, 256, 0, stream>>>(jobs[j].w, jobs[j].p, jobs[j].ntap,
                                        jobs[j].Cin, jobs[j].CinPad, jobs[j].Cout,
                                        jobs[j].total);
    }

    scatter_vox<<<(NV + 255) / 256, 256, 0, stream>>>(feat, coords, x0, m1,
                                                      NV, H1, W1, 4, 32);

    auto pool = [&](const float* in, int Di, int Hi, int Wi, float* outp,
                    int Do, int Ho, int Wo, int wd, int wh, int ww,
                    int sz, int sy, int sx, int pz, int py, int px) {
        int n = Do * Ho * Wo;
        maxpool_m<<<(n + 255) / 256, 256, 0, stream>>>(in, Di, Hi, Wi, outp,
            Do, Ho, Wo, wd, wh, ww, sz, sy, sx, pz, py, px);
    };
    pool(m1, D1, H1, W1, m2, D2, H2, W2, 3,3,3, 2,2,2, 1,1,1);
    pool(m2, D2, H2, W2, m3, D3, H3, W3, 3,3,3, 2,2,2, 1,1,1);
    pool(m3, D3, H3, W3, m4, D4, H4, W4, 3,3,3, 2,2,2, 0,1,1);
    pool(m4, D4, H4, W4, mo, DO, HO, WO, 3,1,1, 2,1,1, 0,0,0);

    auto conv = [&](const _Float16* in, int Di, int Hi, int Wi, int CinPad,
                    _Float16* outp, int Do, int Ho, int Wo, int Cout, int CoutPad,
                    const _Float16* wp, int wi, const float* mask,
                    int kd, int kh, int kw, int sz, int sy, int sx,
                    int pz, int py, int px, int mode, int relu,
                    const _Float16* resid) {
        int nvox = Do * Ho * Wo;
        dim3 g((nvox + 127) / 128, Cout / 16);   // 8 tiles of 16 voxels / block
        int KB = CinPad / 32, ntap = kd * kh * kw;
        size_t wbytes = (size_t)ntap * KB * 1024;
        bool lds = (wbytes <= 64 * 1024);
        unsigned sh = lds ? (unsigned)wbytes : 0;
        const float* sc = Wf(wi + 1);
        const float* bi = Wf(wi + 2);
        if (mode == 0 && kd == 3 && kh == 3 && kw == 3) {
            if (lds)
                conv3d_wmma<3,3,3,0,1><<<g, 256, sh, stream>>>(in, Di, Hi, Wi,
                    CinPad, outp, Do, Ho, Wo, Cout, CoutPad, wp, sc, bi, mask,
                    sz, sy, sx, pz, py, px, relu, resid);
            else
                conv3d_wmma<3,3,3,0,0><<<g, 256, 0, stream>>>(in, Di, Hi, Wi,
                    CinPad, outp, Do, Ho, Wo, Cout, CoutPad, wp, sc, bi, mask,
                    sz, sy, sx, pz, py, px, relu, resid);
        } else if (mode == 1) {
            conv3d_wmma<3,3,3,1,1><<<g, 256, sh, stream>>>(in, Di, Hi, Wi,
                CinPad, outp, Do, Ho, Wo, Cout, CoutPad, wp, sc, bi, mask,
                sz, sy, sx, pz, py, px, relu, resid);
        } else {
            conv3d_wmma<3,1,1,0,1><<<g, 256, sh, stream>>>(in, Di, Hi, Wi,
                CinPad, outp, Do, Ho, Wo, Cout, CoutPad, wp, sc, bi, mask,
                sz, sy, sx, pz, py, px, relu, resid);
        }
    };

    // ---------------- encoder ----------------
    conv(x0, D1,H1,W1,32, ta, D1,H1,W1,16,32, w_in,  IN_W,  m1, 3,3,3, 1,1,1, 1,1,1, 0,1, nullptr);
    conv(ta, D1,H1,W1,32, x1, D1,H1,W1,16,32, w_c1,  C1_W,  m1, 3,3,3, 1,1,1, 1,1,1, 0,1, nullptr);

    conv(x1, D1,H1,W1,32, x2,  D2,H2,W2,32,32, w_c2a, C2A_W, m2, 3,3,3, 2,2,2, 1,1,1, 0,1, nullptr);
    conv(x2, D2,H2,W2,32, t2a, D2,H2,W2,32,32, w_c2b, C2B_W, m2, 3,3,3, 1,1,1, 1,1,1, 0,1, nullptr);
    conv(t2a,D2,H2,W2,32, x2,  D2,H2,W2,32,32, w_c2c, C2C_W, m2, 3,3,3, 1,1,1, 1,1,1, 0,1, nullptr);

    conv(x2, D2,H2,W2,32, x3,  D3,H3,W3,64,64, w_c3a, C3A_W, m3, 3,3,3, 2,2,2, 1,1,1, 0,1, nullptr);
    conv(x3, D3,H3,W3,64, t3a, D3,H3,W3,64,64, w_c3b, C3B_W, m3, 3,3,3, 1,1,1, 1,1,1, 0,1, nullptr);
    conv(t3a,D3,H3,W3,64, x3,  D3,H3,W3,64,64, w_c3c, C3C_W, m3, 3,3,3, 1,1,1, 1,1,1, 0,1, nullptr);

    conv(x3, D3,H3,W3,64, x4,  D4,H4,W4,64,64, w_c4a, C4A_W, m4, 3,3,3, 2,2,2, 0,1,1, 0,1, nullptr);
    conv(x4, D4,H4,W4,64, t4a, D4,H4,W4,64,64, w_c4b, C4B_W, m4, 3,3,3, 1,1,1, 1,1,1, 0,1, nullptr);
    conv(t4a,D4,H4,W4,64, x4,  D4,H4,W4,64,64, w_c4c, C4C_W, m4, 3,3,3, 1,1,1, 1,1,1, 0,1, nullptr);

    conv(x4, D4,H4,W4,64, ob, DO,HO,WO,128,128, w_out, OUT_W, mo, 3,1,1, 2,1,1, 0,0,0, 0,1, nullptr);
    spatial_out<<<(23040 + 255) / 256, 256, 0, stream>>>(ob, (float*)d_out);

    // ---------------- decoder ----------------
    // ur(x4, x4, m4, t4, m4)
    conv(x4, D4,H4,W4,64, t4a, D4,H4,W4,64,64, w_t4a, T4A_W, m4, 3,3,3, 1,1,1, 1,1,1, 0,1, nullptr);
    conv(t4a,D4,H4,W4,64, t4b, D4,H4,W4,64,64, w_t4b, T4B_W, m4, 3,3,3, 1,1,1, 1,1,1, 0,1, x4);
    cat_ch<<<((n4 * 128) + 255) / 256, 256, 0, stream>>>(x4, 64, 64, t4b, 64, 64, cat4, 128, n4);
    conv(cat4,D4,H4,W4,128, t4a, D4,H4,W4,64,64, w_m4, M4_W, m4, 3,3,3, 1,1,1, 1,1,1, 0,1, nullptr);
    red_add<<<((n4 * 64) + 255) / 256, 256, 0, stream>>>(t4a, 64, cat4, 128, u4, 64, n4, 64);
    conv(u4, D4,H4,W4,64, u3d, D3,H3,W3,64,64, w_inv4, INV4_W, m3, 3,3,3, 2,2,2, 0,1,1, 1,1, nullptr);

    // ur(x3, u3d, m3, t3, m3)
    conv(x3, D3,H3,W3,64, t3a, D3,H3,W3,64,64, w_t3a, T3A_W, m3, 3,3,3, 1,1,1, 1,1,1, 0,1, nullptr);
    conv(t3a,D3,H3,W3,64, t3b, D3,H3,W3,64,64, w_t3b, T3B_W, m3, 3,3,3, 1,1,1, 1,1,1, 0,1, x3);
    cat_ch<<<((n3 * 128) + 255) / 256, 256, 0, stream>>>(u3d, 64, 64, t3b, 64, 64, cat3, 128, n3);
    conv(cat3,D3,H3,W3,128, t3a, D3,H3,W3,64,64, w_m3, M3_W, m3, 3,3,3, 1,1,1, 1,1,1, 0,1, nullptr);
    red_add<<<((n3 * 64) + 255) / 256, 256, 0, stream>>>(t3a, 64, cat3, 128, u3, 64, n3, 64);
    conv(u3, D3,H3,W3,64, u2d, D2,H2,W2,32,32, w_inv3, INV3_W, m2, 3,3,3, 2,2,2, 1,1,1, 1,1, nullptr);

    // ur(x2, u2d, m2, t2, m2)
    conv(x2, D2,H2,W2,32, t2a, D2,H2,W2,32,32, w_t2a, T2A_W, m2, 3,3,3, 1,1,1, 1,1,1, 0,1, nullptr);
    conv(t2a,D2,H2,W2,32, t2b, D2,H2,W2,32,32, w_t2b, T2B_W, m2, 3,3,3, 1,1,1, 1,1,1, 0,1, x2);
    cat_ch<<<((n2 * 64) + 255) / 256, 256, 0, stream>>>(u2d, 32, 32, t2b, 32, 32, cat2, 64, n2);
    conv(cat2,D2,H2,W2,64, t2a, D2,H2,W2,32,32, w_m2, M2_W, m2, 3,3,3, 1,1,1, 1,1,1, 0,1, nullptr);
    red_add<<<((n2 * 32) + 255) / 256, 256, 0, stream>>>(t2a, 32, cat2, 64, u2, 32, n2, 32);
    conv(u2, D2,H2,W2,32, u1d, D1,H1,W1,16,32, w_inv2, INV2_W, m1, 3,3,3, 2,2,2, 1,1,1, 1,1, nullptr);

    // ur(x1, u1d, m1, t1, m1)
    conv(x1, D1,H1,W1,32, ta, D1,H1,W1,16,32, w_t1a, T1A_W, m1, 3,3,3, 1,1,1, 1,1,1, 0,1, nullptr);
    conv(ta, D1,H1,W1,32, tb, D1,H1,W1,16,32, w_t1b, T1B_W, m1, 3,3,3, 1,1,1, 1,1,1, 0,1, x1);
    cat_ch<<<((n1 * 32) + 255) / 256, 256, 0, stream>>>(u1d, 32, 16, tb, 32, 16, x0, 32, n1);
    conv(x0, D1,H1,W1,32, ta, D1,H1,W1,16,32, w_m1, M1_W, m1, 3,3,3, 1,1,1, 1,1,1, 0,1, nullptr);
    red_add<<<((n1 * 16) + 255) / 256, 256, 0, stream>>>(ta, 32, x0, 32, tb, 32, n1, 16);
    conv(tb, D1,H1,W1,32, ta, D1,H1,W1,16,32, w_c5, C5_W, m1, 3,3,3, 1,1,1, 1,1,1, 0,1, nullptr);

    float* dout = (float*)d_out;
    heads<<<(NV + 255) / 256, 256, 0, stream>>>(ta, 32, coords,
        Wf(CLSW), Wf(CLSB), Wf(REGW), Wf(REGB),
        dout + 23040, dout + 53040, dout + 143040, NV, H1, W1);
}
